// SelfAttentionConv_6657199308858
// MI455X (gfx1250) — matrix operand order repaired and checked by reading, not compiled, not run on hardware
//
#include <hip/hip_runtime.h>
#include <stdint.h>
#include <stddef.h>

// ---------------------------------------------------------------------------
// SelfAttentionConv for MI455X (gfx1250), wave32 + WMMA bf16 + TDM.
// B=4, T=2048, K=128, H=8, KS=3.
//
// Pipeline (all on `stream`):
//   0) prep:    reorder conv/unify weights to bf16 [c][f] layouts  (ws)
//   1) conv:    q/k/v projections as WMMA GEMMs, scatter into the
//               "faithful torch reshape" space  n=bi*8+t/256, s=(t%256)*8+h
//   2) attn:    flash attention with online softmax over the reshaped space
//               (causal tril in s-space); K tiles staged to LDS via the
//               Tensor Data Mover (tensor_load_to_lds + s_wait_tensorcnt),
//               O overwrites the Q buffer
//   3) unify:   out = O @ w_unify^T + b_unify  (fp32 output)
//
// Workspace map (bytes), total ~50.3 MB:
//   [0,16M)   Qp  bf16 (32,2048,128)  -> aliased as attention output O
//   [16,32M)  Kp  bf16
//   [32,48M)  Vp  bf16
//   [48M..)   Wq'' (1024x384), Wk'' (1024x384), Wv'' (1024x128), Wu'' (128x1024)
// ---------------------------------------------------------------------------

typedef unsigned short u16;
typedef __bf16 bf16;
typedef bf16  v16bf __attribute__((ext_vector_type(16)));
typedef float v8f   __attribute__((ext_vector_type(8)));
typedef unsigned uv4 __attribute__((ext_vector_type(4)));
typedef float   fv4 __attribute__((ext_vector_type(4)));

// TDM descriptor group types (per CDNA5_HIP.md probe):
typedef unsigned uint32x4 __attribute__((ext_vector_type(4)));
typedef int      int32x8  __attribute__((ext_vector_type(8)));
typedef int      int32x4  __attribute__((ext_vector_type(4)));

#if defined(__gfx1250__) && __has_builtin(__builtin_amdgcn_tensor_load_to_lds) && \
    __has_builtin(__builtin_amdgcn_s_wait_tensorcnt)
#define HAVE_TDM 1
#else
#define HAVE_TDM 0
#endif

#define WMMA_BF16(a, b, c) \
  __builtin_amdgcn_wmma_f32_16x16x32_bf16(false, (a), false, (b), (short)0, (c), false, false)

__device__ __forceinline__ u16 f2bf(float x) {
  union { float f; unsigned u; } v; v.f = x;
  unsigned r = v.u + 0x7FFFu + ((v.u >> 16) & 1u);   // round-to-nearest-even
  return (u16)(r >> 16);
}

union Frag16 { v16bf v; u16 us[16]; uv4 q[2]; };

// B fragment (32x16, k' = g*16 + e): 16 contiguous bf16 per lane.
__device__ __forceinline__ v16bf load_b16c(const u16* p) {
  Frag16 f;
  f.q[0] = *(const uv4*)(p);
  f.q[1] = *(const uv4*)(p + 8);
  return f.v;
}
// A fragment (16x32): two 8-bf16 runs per lane (k = g*16+e and g*16+8+e).
__device__ __forceinline__ v16bf load_a16(const u16* row, int o1, int o2) {
  Frag16 f;
  f.q[0] = *(const uv4*)(row + o1);
  f.q[1] = *(const uv4*)(row + o2);
  return f.v;
}

__device__ __forceinline__ void pack8(u16* d, const float* s) {
  fv4 a = *(const fv4*)(s);
  fv4 b = *(const fv4*)(s + 4);
  d[0] = f2bf(a.x); d[1] = f2bf(a.y); d[2] = f2bf(a.z); d[3] = f2bf(a.w);
  d[4] = f2bf(b.x); d[5] = f2bf(b.y); d[6] = f2bf(b.z); d[7] = f2bf(b.w);
}

#define INV_S4 0.2973017787506803f   // 1 / 128^0.25

// ---------------------------------------------------------------------------
// Kernel 0: weight prep -> bf16, layout [c_out][f] so B-frags are contiguous.
// ---------------------------------------------------------------------------
__global__ __launch_bounds__(256) void prep_kernel(
    const float* __restrict__ wq, const float* __restrict__ wk,
    const float* __restrict__ wv, const float* __restrict__ wu,
    u16* __restrict__ Wq, u16* __restrict__ Wk,
    u16* __restrict__ Wv, u16* __restrict__ Wu) {
  unsigned idx = blockIdx.x * 256u + threadIdx.x;
  if (idx < 1024u * 384u) {
    unsigned c = idx / 384u, f = idx % 384u;
    unsigned w = f >> 7, i = f & 127u;
    Wq[idx] = f2bf(wq[(c * 128u + i) * 3u + w]);
    Wk[idx] = f2bf(wk[(c * 128u + i) * 3u + w]);
  }
  if (idx < 131072u) {
    Wv[idx] = f2bf(wv[idx]);
    Wu[idx] = f2bf(wu[idx]);
  }
}

// ---------------------------------------------------------------------------
// Kernel 1: conv q/k/v as WMMA GEMM, scatter into reshaped (n, s, kk) space.
// One wave = one 16-row m-tile (16 consecutive t for one batch).
// ---------------------------------------------------------------------------
__global__ __launch_bounds__(256) void conv_qkv_kernel(
    const float* __restrict__ x,
    const float* __restrict__ bq, const float* __restrict__ bk,
    const u16* __restrict__ Wq, const u16* __restrict__ Wk,
    const u16* __restrict__ Wv,
    u16* __restrict__ Qp, u16* __restrict__ Kp, u16* __restrict__ Vp) {
  const int tid = threadIdx.x;
  const int wv_ = tid >> 5;
  const int lane = tid & 31;
  const int nidx = lane & 15;
  const int g = lane >> 4;

  const int mtile = blockIdx.x * 8 + wv_;
  const int mbase = mtile * 16;
  const int bi = mbase >> 11;
  const int tbase = mbase & 2047;
  const int t = tbase + nidx;              // A-fragment row for this lane

  // Build 12 A fragments (f = w*128 + i, causal taps t-2+w, zero padded).
  v16bf af[12];
  #pragma unroll
  for (int kc = 0; kc < 12; ++kc) {
    const int w = kc >> 2;
    const int i0 = (kc & 3) * 32;
    const int tsrc = t - 2 + w;
    Frag16 f;
    if (tsrc >= 0) {
      const float* xr = x + ((size_t)(bi * 2048 + tsrc)) * 128;
      pack8(f.us,     xr + i0 + g * 16);
      pack8(f.us + 8, xr + i0 + g * 16 + 8);
    } else {
      #pragma unroll
      for (int e = 0; e < 16; ++e) f.us[e] = 0;
    }
    af[kc] = f.v;
  }

  for (int nt = 0; nt < 64; ++nt) {
    const int c = nt * 16 + nidx;          // output channel for this lane
    const int kk = c >> 3;
    const int hh = c & 7;

    // ---- Q ----
    v8f qa = {};
    #pragma unroll
    for (int kc = 0; kc < 12; ++kc) {
      v16bf b = load_b16c(Wq + (size_t)c * 384 + kc * 32 + g * 16);
      qa = WMMA_BF16(af[kc], b, qa);
    }
    const float biasq = bq[c];
    #pragma unroll
    for (int r = 0; r < 8; ++r) {
      const int tr = tbase + r + 8 * g;
      const int na = bi * 8 + (tr >> 8);
      const int s  = (tr & 255) * 8 + hh;
      Qp[((size_t)na * 2048 + s) * 128 + kk] = f2bf((qa[r] + biasq) * INV_S4);
    }

    // ---- K ----
    v8f ka = {};
    #pragma unroll
    for (int kc = 0; kc < 12; ++kc) {
      v16bf b = load_b16c(Wk + (size_t)c * 384 + kc * 32 + g * 16);
      ka = WMMA_BF16(af[kc], b, ka);
    }
    const float biask = bk[c];
    #pragma unroll
    for (int r = 0; r < 8; ++r) {
      const int tr = tbase + r + 8 * g;
      const int na = bi * 8 + (tr >> 8);
      const int s  = (tr & 255) * 8 + hh;
      Kp[((size_t)na * 2048 + s) * 128 + kk] = f2bf((ka[r] + biask) * INV_S4);
    }

    // ---- V (tap w=2 only => A fragments 8..11) ----
    v8f va = {};
    #pragma unroll
    for (int kc = 0; kc < 4; ++kc) {
      v16bf b = load_b16c(Wv + (size_t)c * 128 + kc * 32 + g * 16);
      va = WMMA_BF16(af[8 + kc], b, va);
    }
    #pragma unroll
    for (int r = 0; r < 8; ++r) {
      const int tr = tbase + r + 8 * g;
      const int na = bi * 8 + (tr >> 8);
      const int s  = (tr & 255) * 8 + hh;
      Vp[((size_t)na * 2048 + s) * 128 + kk] = f2bf(va[r]);
    }
  }
}

// ---------------------------------------------------------------------------
// Kernel 2: causal flash attention over the reshaped space.
// Block = (n, 128-row block). Wave = 16 rows. K tile (32x128, contiguous 8KB)
// is DMA'd into LDS by the Tensor Data Mover (wave 0 issues one descriptor,
// tracked by TENSORcnt); V tile is transposed into LDS manually so PV B-frags
// are contiguous. O overwrites the Q buffer.
// ---------------------------------------------------------------------------
__global__ __launch_bounds__(256) void attn_kernel(
    const u16* __restrict__ Kp, const u16* __restrict__ Vp,
    u16* QOp) {
  __shared__ __align__(16) u16 Kt[32][128];     // K tile, row-major [j][d]
  __shared__ __align__(16) u16 Vt[128][32];     // V tile transposed [d][j]
  __shared__ __align__(16) u16 Pst[8][16][32];  // per-wave P staging

  const int tid = threadIdx.x;
  const int wv_ = tid >> 5;
  const int lane = tid & 31;
  const int nidx = lane & 15;
  const int g = lane >> 4;

  const int n  = blockIdx.x >> 4;
  const int rb = blockIdx.x & 15;
  const int row0 = rb * 128 + wv_ * 16;

  // Preload Q tile (16 x 128) as 4 A fragments.
  const u16* Qrow = QOp + ((size_t)n * 2048 + row0 + nidx) * 128;
  v16bf qf[4];
  #pragma unroll
  for (int c = 0; c < 4; ++c)
    qf[c] = load_a16(Qrow, c * 32 + g * 16, c * 32 + g * 16 + 8);

  float m_i[8], l_i[8];
  v8f oacc[8];
  #pragma unroll
  for (int r = 0; r < 8; ++r) { m_i[r] = -1e30f; l_i[r] = 0.0f; }
  #pragma unroll
  for (int nt = 0; nt < 8; ++nt) { v8f z = {}; oacc[nt] = z; }

  const int nblk = rb * 4 + 4;            // col blocks of 32 up to the diagonal
  const int ldrow = tid >> 3;             // 0..31 : tile row loaded by this thread
  const int ldc   = (tid & 7) * 16;       // 16-feature chunk

#if HAVE_TDM
  // Invariant D# pieces for the 8KB contiguous K-tile copy (data_size=3: 8B
  // units, 1-D tile of 1024 units, tensor_dim0 == tile_dim0 -> no OOB).
  const unsigned ldsKt =
      (unsigned)(unsigned long)(__attribute__((address_space(3))) u16*)&Kt[0][0];
  int32x8 tdm_g1;
  tdm_g1[0] = (int)(3u << 16);            // data_size = 3 (8 bytes)
  tdm_g1[1] = (int)(1024u << 16);         // tensor_dim0[15:0]  (=1024)
  tdm_g1[2] = (int)(1u << 16);            // tensor_dim0 hi=0 | tensor_dim1 lo=1
  tdm_g1[3] = (int)(1024u << 16);         // tensor_dim1 hi=0 | tile_dim0=1024
  tdm_g1[4] = 1;                          // tile_dim1=1, tile_dim2=0
  tdm_g1[5] = 1024;                       // tensor_dim0_stride lo32
  tdm_g1[6] = 0;                          // stride hi | tensor_dim1_stride lo
  tdm_g1[7] = 0;
  const int32x4 tdm_g2 = {0, 0, 0, 0};
  const int32x4 tdm_g3 = {0, 0, 0, 0};
#endif

  for (int jb = 0; jb < nblk; ++jb) {
    const int j0 = jb * 32;
    __syncthreads();
    {
      const size_t gbase = ((size_t)n * 2048 + j0 + ldrow) * 128 + ldc;
#if HAVE_TDM
      if (wv_ == 0) {                     // one TDM op moves the whole K tile
        const unsigned long long gaddr =
            (unsigned long long)(const void*)(Kp + ((size_t)n * 2048 + j0) * 128);
        uint32x4 g0;
        g0[0] = 1u;                                       // count = 1
        g0[1] = ldsKt;                                    // lds_addr
        g0[2] = (unsigned)(gaddr & 0xFFFFFFFFull);        // global_addr lo
        g0[3] = (unsigned)((gaddr >> 32) & 0x01FFFFFFull) // global_addr hi
                | (2u << 30);                             // type = 2 (image)
#if __clang_major__ >= 23
        const int32x8 z8 = {0, 0, 0, 0, 0, 0, 0, 0};
        __builtin_amdgcn_tensor_load_to_lds(g0, tdm_g1, tdm_g2, tdm_g3, z8, 0);
#else
        __builtin_amdgcn_tensor_load_to_lds(g0, tdm_g1, tdm_g2, tdm_g3, 0);
#endif
      }
#else
      const u16* gk = Kp + gbase;
      *(uv4*)&Kt[ldrow][ldc]     = *(const uv4*)gk;
      *(uv4*)&Kt[ldrow][ldc + 8] = *(const uv4*)(gk + 8);
#endif
      // V tile: load 16 features and transpose into LDS.
      Frag16 fv;
      const u16* gv = Vp + gbase;
      fv.q[0] = *(const uv4*)gv;
      fv.q[1] = *(const uv4*)(gv + 8);
      #pragma unroll
      for (int e = 0; e < 16; ++e) Vt[ldc + e][ldrow] = fv.us[e];
      if (jb + 1 < nblk) {                       // prefetch next tile
        __builtin_prefetch(Kp + gbase + 32 * 128, 0, 1);
        __builtin_prefetch(Vp + gbase + 32 * 128, 0, 1);
      }
#if HAVE_TDM
      if (wv_ == 0) __builtin_amdgcn_s_wait_tensorcnt(0);  // drain TDM before
#endif                                                     // barrier arrival
    }
    __syncthreads();

    // S = Q K^T : two 16x16 tiles (columns j0..j0+15 / j0+16..j0+31)
    v8f s0 = {}, s1 = {};
    #pragma unroll
    for (int c = 0; c < 4; ++c) {
      v16bf b0 = load_b16c(&Kt[nidx][c * 32 + g * 16]);
      s0 = WMMA_BF16(qf[c], b0, s0);
      v16bf b1 = load_b16c(&Kt[16 + nidx][c * 32 + g * 16]);
      s1 = WMMA_BF16(qf[c], b1, s1);
    }

    // Online softmax (row stats reduced across each 16-lane half).
    float alpha[8];
    #pragma unroll
    for (int r = 0; r < 8; ++r) {
      const int rowg = row0 + r + 8 * g;
      float a0 = (j0 + nidx      <= rowg) ? s0[r] : -1e30f;
      float a1 = (j0 + 16 + nidx <= rowg) ? s1[r] : -1e30f;
      float mx = fmaxf(a0, a1);
      mx = fmaxf(mx, __shfl_xor(mx, 1, 32));
      mx = fmaxf(mx, __shfl_xor(mx, 2, 32));
      mx = fmaxf(mx, __shfl_xor(mx, 4, 32));
      mx = fmaxf(mx, __shfl_xor(mx, 8, 32));
      const float mnew = fmaxf(m_i[r], mx);
      const float al = __expf(m_i[r] - mnew);
      const float p0 = __expf(a0 - mnew);
      const float p1 = __expf(a1 - mnew);
      float rs = p0 + p1;
      rs += __shfl_xor(rs, 1, 32);
      rs += __shfl_xor(rs, 2, 32);
      rs += __shfl_xor(rs, 4, 32);
      rs += __shfl_xor(rs, 8, 32);
      l_i[r] = l_i[r] * al + rs;
      m_i[r] = mnew;
      alpha[r] = al;
      Pst[wv_][r + 8 * g][nidx]      = f2bf(p0);
      Pst[wv_][r + 8 * g][16 + nidx] = f2bf(p1);
    }
    #pragma unroll
    for (int nt = 0; nt < 8; ++nt)
      #pragma unroll
      for (int r = 0; r < 8; ++r) oacc[nt][r] *= alpha[r];

    // P (C-layout) -> A-layout through per-wave LDS staging; same-wave DS
    // ops are in order, so no barrier is needed.
    v16bf pa = load_a16(&Pst[wv_][nidx][0], g * 16, g * 16 + 8);
    #pragma unroll
    for (int nt = 0; nt < 8; ++nt) {
      v16bf bvv = load_b16c(&Vt[nt * 16 + nidx][g * 16]);
      oacc[nt] = WMMA_BF16(pa, bvv, oacc[nt]);
    }
  }

  // Normalize and store O (overwrites this block's own Q rows).
  float invl[8];
  #pragma unroll
  for (int r = 0; r < 8; ++r) invl[r] = 1.0f / l_i[r];
  #pragma unroll
  for (int nt = 0; nt < 8; ++nt)
    #pragma unroll
    for (int r = 0; r < 8; ++r)
      QOp[((size_t)n * 2048 + row0 + r + 8 * g) * 128 + nt * 16 + nidx] =
          f2bf(oacc[nt][r] * invl[r]);
}

// ---------------------------------------------------------------------------
// Kernel 3: unify GEMM  out[bi,t,ko] = sum_f O[bi*8+f/128, t, f%128]*Wu[ko,f] + bu
// ---------------------------------------------------------------------------
__global__ __launch_bounds__(256) void unify_kernel(
    const u16* __restrict__ Op, const u16* __restrict__ Wu,
    const float* __restrict__ bu, float* __restrict__ out) {
  const int tid = threadIdx.x;
  const int wv_ = tid >> 5;
  const int lane = tid & 31;
  const int nidx = lane & 15;
  const int g = lane >> 4;

  const int mtile = blockIdx.x * 8 + wv_;
  const int mbase = mtile * 16;
  const int bi = mbase >> 11;
  const int tbase = mbase & 2047;
  const int t = tbase + nidx;

  v8f acc[8];
  #pragma unroll
  for (int nt = 0; nt < 8; ++nt) { v8f z = {}; acc[nt] = z; }

  for (int kc = 0; kc < 32; ++kc) {
    const int f0 = kc * 32;
    const int h  = f0 >> 7;
    const int k0 = f0 & 127;
    const u16* arow = Op + ((size_t)(bi * 8 + h) * 2048 + t) * 128;
    v16bf a = load_a16(arow, k0 + g * 16, k0 + g * 16 + 8);
    #pragma unroll
    for (int nt = 0; nt < 8; ++nt) {
      v16bf b = load_b16c(Wu + (size_t)(nt * 16 + nidx) * 1024 + f0 + g * 16);
      acc[nt] = WMMA_BF16(a, b, acc[nt]);
    }
  }

  #pragma unroll
  for (int nt = 0; nt < 8; ++nt) {
    const int ko = nt * 16 + nidx;
    const float bb = bu[ko];
    #pragma unroll
    for (int r = 0; r < 8; ++r) {
      const int tr = tbase + r + 8 * g;
      out[((size_t)bi * 2048 + tr) * 128 + ko] = acc[nt][r] + bb;
    }
  }
}

// ---------------------------------------------------------------------------
extern "C" void kernel_launch(void* const* d_in, const int* in_sizes, int n_in,
                              void* d_out, int out_size, void* d_ws, size_t ws_size,
                              hipStream_t stream) {
  (void)in_sizes; (void)n_in; (void)out_size; (void)ws_size;
  const float* x  = (const float*)d_in[0];
  const float* wq = (const float*)d_in[1];
  const float* bq = (const float*)d_in[2];
  const float* wk = (const float*)d_in[3];
  const float* bk = (const float*)d_in[4];
  const float* wv = (const float*)d_in[5];
  const float* wu = (const float*)d_in[6];
  const float* bu = (const float*)d_in[7];
  float* out = (float*)d_out;

  char* ws = (char*)d_ws;
  u16* Qp = (u16*)(ws);                                  // 16 MB (also O)
  u16* Kp = (u16*)(ws + (size_t)16 * 1024 * 1024);       // 16 MB
  u16* Vp = (u16*)(ws + (size_t)32 * 1024 * 1024);       // 16 MB
  u16* Wq = (u16*)(ws + (size_t)48 * 1024 * 1024);       // 1024*384
  u16* Wk = Wq + 393216;                                 // 1024*384
  u16* Wv = Wk + 393216;                                 // 1024*128
  u16* Wu = Wv + 131072;                                 // 128*1024

  prep_kernel<<<1536, 256, 0, stream>>>(wq, wk, wv, wu, Wq, Wk, Wv, Wu);
  conv_qkv_kernel<<<64, 256, 0, stream>>>(x, bq, bk, Wq, Wk, Wv, Qp, Kp, Vp);
  attn_kernel<<<512, 256, 0, stream>>>(Kp, Vp, Qp);
  unify_kernel<<<64, 256, 0, stream>>>(Qp, Wu, bu, out);
}
